// GATClassifier_33775622816321
// MI455X (gfx1250) — compile-verified, matrix operand order checked by reading, not run on hardware
//
#include <hip/hip_runtime.h>
#include <hip/hip_bf16.h>
#include <math.h>

// ---------------- types for WMMA ----------------
typedef __attribute__((ext_vector_type(16))) __bf16 v16bf;
typedef __attribute__((ext_vector_type(8)))  float  v8f;

#define HEADS 4
#define HID   32
#define HC    128      // HEADS*HID
#define EDIM  16
#define NG    64
#define OUTD  2

// order-preserving float<->uint encoding for atomic max
__device__ __forceinline__ unsigned encf(float f) {
  unsigned u = __float_as_uint(f);
  return (u & 0x80000000u) ? ~u : (u | 0x80000000u);
}
__device__ __forceinline__ float decf(unsigned e) {
  unsigned u = (e & 0x80000000u) ? (e & 0x7FFFFFFFu) : ~e;
  return __uint_as_float(u);
}
#define ENC_NEG_INF 0x007FFFFFu   // encf(-inf)

__device__ __forceinline__ unsigned short f2bf(float f) {
  unsigned u = __float_as_uint(f);
  unsigned r = u + 0x7FFFu + ((u >> 16) & 1u);   // round-to-nearest-even
  return (unsigned short)(r >> 16);
}

// ---------------- utility fills ----------------
__global__ void fill_f32(float* p, float v, int n) {
  int i = blockIdx.x * blockDim.x + threadIdx.x;
  if (i < n) p[i] = v;
}
__global__ void fill_u32(unsigned* p, unsigned v, int n) {
  int i = blockIdx.x * blockDim.x + threadIdx.x;
  if (i < n) p[i] = v;
}

// ---------------- edge_attr column sums (for self-loop mean) ----------------
__global__ void ea_sum_kernel(const float* ea, float* sum16, int E) {
  __shared__ float s[16];
  if (threadIdx.x < 16) s[threadIdx.x] = 0.f;
  __syncthreads();
  size_t total = (size_t)E * 16;
  for (size_t i = (size_t)blockIdx.x * blockDim.x + threadIdx.x; i < total;
       i += (size_t)gridDim.x * blockDim.x)
    atomicAdd(&s[i & 15], ea[i]);
  __syncthreads();
  if (threadIdx.x < 16) atomicAdd(&sum16[threadIdx.x], s[threadIdx.x]);
}

// veff[k][h] = sum_c We[k,h*32+c]*ae[h,c]; aloop[h] = mean_ea . veff[:,h]
__global__ void layer_prep_kernel(const float* sum16, const float* We, const float* ae,
                                  float* veff, float* aloop, float Einv) {
  int t = threadIdx.x;
  if (t < 64) {
    int k = t >> 2, hh = t & 3;
    float v = 0.f;
    for (int c = 0; c < HID; ++c) v += We[k * HC + hh * HID + c] * ae[hh * HID + c];
    veff[t] = v;
  }
  __syncthreads();
  if (t < 4) {
    float v = 0.f;
    for (int k = 0; k < EDIM; ++k) v += (sum16[k] * Einv) * veff[k * 4 + t];
    aloop[t] = v;
  }
}

// ---------------- precision conversion ----------------
__global__ void cvt_bf16_kernel(const float* in, unsigned short* out, int n) {
  int i = blockIdx.x * blockDim.x + threadIdx.x;
  if (i < n) out[i] = f2bf(in[i]);
}
// W [K,128] f32 -> Wt [128,K] bf16 (column-major for fragment loads)
__global__ void transpose_w_kernel(const float* W, unsigned short* Wt, int K) {
  int i = blockIdx.x * blockDim.x + threadIdx.x;
  if (i < 128 * K) {
    int n = i / K, k = i % K;
    Wt[i] = f2bf(W[(size_t)k * 128 + n]);
  }
}

// ---------------- WMMA GEMM: C[M,128] = A[M,K](bf16) @ B[K,128] (Bt col-major bf16) ----
__device__ __forceinline__ v16bf load_frag16(const unsigned short* p) {
  v16bf r;
  uint4 a = *reinterpret_cast<const uint4*>(p);
  uint4 b = *reinterpret_cast<const uint4*>(p + 16);
  __builtin_memcpy(&r, &a, 16);
  __builtin_memcpy(reinterpret_cast<char*>(&r) + 16, &b, 16);
  return r;
}

__global__ void gemm_wmma_kernel(const unsigned short* __restrict__ A,
                                 const unsigned short* __restrict__ Bt,
                                 float* __restrict__ C, int M, int K) {
  int wave = (blockIdx.x * blockDim.x + threadIdx.x) >> 5;
  int lane = threadIdx.x & 31;
  int row0 = wave * 16;
  if (row0 >= M) return;                 // uniform per wave
  int hi  = lane >> 4;                   // 0 or 1
  int m   = lane & 15;
  int off = hi * 8;                      // K sub-offset per ISA 16-bit A/B layout
  v8f acc[8] = {};
  const unsigned short* arow  = A  + (size_t)(row0 + m) * K + off;
  const unsigned short* bbase = Bt + (size_t)m * K + off;
  for (int kk = 0; kk < K; kk += 32) {
    // hoist ALL fragment loads for this K-step: one load clause, one wait,
    // then 8 independent back-to-back WMMAs (no D->A/B RAW hazards)
    v16bf af = load_frag16(arow + kk);
    v16bf bfg[8];
#pragma unroll
    for (int nt = 0; nt < 8; ++nt)
      bfg[nt] = load_frag16(bbase + (size_t)nt * 16 * K + kk);
#pragma unroll
    for (int nt = 0; nt < 8; ++nt)
      acc[nt] = __builtin_amdgcn_wmma_f32_16x16x32_bf16(
          false, af, false, bfg[nt], (short)0, acc[nt], false, false);
  }
#pragma unroll
  for (int nt = 0; nt < 8; ++nt)
#pragma unroll
    for (int r = 0; r < 8; ++r)
      C[(size_t)(row0 + r + hi * 8) * HC + nt * 16 + m] = acc[nt][r];
}

// ---------------- node attention scalars ----------------
__global__ void attn_node_kernel(const float* __restrict__ h, const float* __restrict__ as_,
                                 const float* __restrict__ ad_, float* asrc, float* adst, int Nn) {
  int i = blockIdx.x * blockDim.x + threadIdx.x;
  if (i >= Nn * HEADS) return;
  int n = i >> 2, hh = i & 3;
  const float* hp = h + (size_t)n * HC + hh * HID;
  float s = 0.f, d = 0.f;
  for (int c = 0; c < HID; ++c) {
    float v = hp[c];
    s += v * as_[hh * HID + c];
    d += v * ad_[hh * HID + c];
  }
  asrc[i] = s;
  adst[i] = d;
}

// ---------------- edge pass 1: alpha + segment max ----------------
__global__ void edge_alpha_kernel(const int* src, const int* dst, const float* ea,
                                  const float* veff, const float* aloop,
                                  const float* asrc, const float* adst,
                                  float* alpha, unsigned* mx, int E, int Nn) {
  __shared__ float sv[64];
  __shared__ float sl[4];
  if (threadIdx.x < 64) sv[threadIdx.x] = veff[threadIdx.x];
  if (threadIdx.x < 4)  sl[threadIdx.x] = aloop[threadIdx.x];
  __syncthreads();
  int e = blockIdx.x * blockDim.x + threadIdx.x;
  int ET = E + Nn;
  if (e >= ET) return;
  int s, d;
  float aeh[4];
  if (e < E) {
    s = src[e]; d = dst[e];
    float v0 = 0.f, v1 = 0.f, v2 = 0.f, v3 = 0.f;
    const float* er = ea + (size_t)e * EDIM;
    for (int k = 0; k < EDIM; ++k) {
      float x = er[k];
      v0 += x * sv[k * 4 + 0]; v1 += x * sv[k * 4 + 1];
      v2 += x * sv[k * 4 + 2]; v3 += x * sv[k * 4 + 3];
    }
    aeh[0] = v0; aeh[1] = v1; aeh[2] = v2; aeh[3] = v3;
  } else {
    s = d = e - E;
    aeh[0] = sl[0]; aeh[1] = sl[1]; aeh[2] = sl[2]; aeh[3] = sl[3];
  }
  for (int hh = 0; hh < HEADS; ++hh) {
    float a = asrc[s * HEADS + hh] + adst[d * HEADS + hh] + aeh[hh];
    a = a > 0.f ? a : 0.2f * a;          // leaky_relu(0.2)
    alpha[(size_t)e * HEADS + hh] = a;
    atomicMax(&mx[d * HEADS + hh], encf(a));
  }
}

// ---------------- edge pass 2: exp + denominator ----------------
__global__ void edge_exp_kernel(const int* dst, float* alpha, const unsigned* mx,
                                float* den, int E, int Nn) {
  int e = blockIdx.x * blockDim.x + threadIdx.x;
  int ET = E + Nn;
  if (e >= ET) return;
  int d = (e < E) ? dst[e] : (e - E);
  for (int hh = 0; hh < HEADS; ++hh) {
    float a = alpha[(size_t)e * HEADS + hh];
    float m = decf(mx[d * HEADS + hh]);
    float w = __expf(a - m);
    alpha[(size_t)e * HEADS + hh] = w;   // becomes ex
    atomicAdd(&den[d * HEADS + hh], w);
  }
}

// ---------------- edge pass 3: aggregate (one wave per edge) ----------------
__global__ void edge_aggr_kernel(const int* src, const int* dst, const float* ex,
                                 const float* den, const float* __restrict__ h,
                                 float* acc, int E, int Nn) {
  int e = blockIdx.x * 8 + (threadIdx.x >> 5);
  int lane = threadIdx.x & 31;
  int ET = E + Nn;
  if (e >= ET) return;                   // uniform per wave
  int s, d;
  if (e < E) { s = src[e]; d = dst[e]; } else { s = d = e - E; }
  int hh = lane >> 3;                    // channel block lane*4 -> head
  float a = ex[(size_t)e * HEADS + hh] / (den[d * HEADS + hh] + 1e-16f);
  float4 hv = *reinterpret_cast<const float4*>(h + (size_t)s * HC + lane * 4);
  float* o = acc + (size_t)d * HC + lane * 4;
  atomicAdd(o + 0, hv.x * a);
  atomicAdd(o + 1, hv.y * a);
  atomicAdd(o + 2, hv.z * a);
  atomicAdd(o + 3, hv.w * a);
}

// ---------------- post: bias (+head-mean) + BN statistics ----------------
__global__ void post_y_stats_kernel(const float* acc, const float* bias, float* y,
                                    float* bnsum, float* bnsq, int Nn, int concat) {
  float s = 0.f, sq = 0.f;
  int c;
  if (concat) {
    c = threadIdx.x;                     // 128 channels
    for (int n = blockIdx.x; n < Nn; n += gridDim.x) {
      float v = acc[(size_t)n * HC + c] + bias[c];
      y[(size_t)n * HC + c] = v;
      s += v; sq += v * v;
    }
  } else {
    c = threadIdx.x & 31;
    int sub = threadIdx.x >> 5;          // 4 nodes per block-iteration
    for (int n = blockIdx.x * 4 + sub; n < Nn; n += gridDim.x * 4) {
      const float* p = acc + (size_t)n * HC + c;
      float v = 0.25f * (p[0] + p[32] + p[64] + p[96]) + bias[c];
      y[(size_t)n * HID + c] = v;
      s += v; sq += v * v;
    }
  }
  atomicAdd(&bnsum[c], s);
  atomicAdd(&bnsq[c], sq);
}

__global__ void bn_finalize_kernel(const float* bnsum, const float* bnsq,
                                   const float* g, const float* be,
                                   float* sc, float* sh, int Nn, int C) {
  int c = threadIdx.x;
  if (c < C) {
    float mu  = bnsum[c] / (float)Nn;
    float var = bnsq[c] / (float)Nn - mu * mu;
    float s   = g[c] * rsqrtf(var + 1e-5f);
    sc[c] = s;
    sh[c] = be[c] - mu * s;
  }
}

__global__ void bn_elu_cvt_kernel(const float* y, const float* sc, const float* sh,
                                  float* xn, unsigned short* xb, int Nn, int C) {
  int i = blockIdx.x * blockDim.x + threadIdx.x;
  if (i >= Nn * C) return;
  int c = i % C;
  float v = y[i] * sc[c] + sh[c];
  float e = v > 0.f ? v : expm1f(v);     // ELU(alpha=1)
  xn[i] = e;
  xb[i] = f2bf(e);
}

// ---------------- pooling ----------------
__global__ void gate_max_kernel(const float* x, const float* gw, const float* gb,
                                const int* batch, float* gate, unsigned* gmax, int Nn) {
  int n = blockIdx.x * blockDim.x + threadIdx.x;
  if (n >= Nn) return;
  float s = gb[0];
  const float* xp = x + (size_t)n * HID;
  for (int c = 0; c < HID; ++c) s += xp[c] * gw[c];
  gate[n] = s;
  atomicMax(&gmax[batch[n]], encf(s));
}

__global__ void gate_exp_kernel(const float* gate, const int* batch, const unsigned* gmax,
                                float* wexp, float* gden, int Nn) {
  int n = blockIdx.x * blockDim.x + threadIdx.x;
  if (n >= Nn) return;
  int b = batch[n];
  float w = __expf(gate[n] - decf(gmax[b]));
  wexp[n] = w;
  atomicAdd(&gden[b], w);
}

__global__ void pool_kernel(const float* x, const float* wexp, const float* gden,
                            const int* batch, float* pooled, int Nn) {
  int i = blockIdx.x * blockDim.x + threadIdx.x;
  if (i >= Nn * HID) return;
  int n = i >> 5, c = i & 31;
  int b = batch[n];
  float w = wexp[n] / (gden[b] + 1e-16f);
  atomicAdd(&pooled[b * HID + c], w * x[(size_t)n * HID + c]);
}

__global__ void fc_kernel(const float* pooled, const float* fcW, const float* fcb,
                          float* out) {
  int t = threadIdx.x;
  if (t >= NG * OUTD) return;
  int g = t / OUTD, o = t % OUTD;
  float s = fcb[o];
  for (int c = 0; c < HID; ++c) s += pooled[g * HID + c] * fcW[c * OUTD + o];
  out[g * OUTD + o] = s;
}

// =====================================================================
extern "C" void kernel_launch(void* const* d_in, const int* in_sizes, int n_in,
                              void* d_out, int out_size, void* d_ws, size_t ws_size,
                              hipStream_t stream) {
  const float* x_in   = (const float*)d_in[0];
  const int*   src    = (const int*)d_in[1];
  const int*   dst    = (const int*)d_in[2];
  const float* eattr  = (const float*)d_in[3];
  const int*   batch  = (const int*)d_in[4];
  const float* W[3]   = { (const float*)d_in[5],  (const float*)d_in[11], (const float*)d_in[17] };
  const float* As[3]  = { (const float*)d_in[6],  (const float*)d_in[12], (const float*)d_in[18] };
  const float* Ad[3]  = { (const float*)d_in[7],  (const float*)d_in[13], (const float*)d_in[19] };
  const float* Wev[3] = { (const float*)d_in[8],  (const float*)d_in[14], (const float*)d_in[20] };
  const float* Ae[3]  = { (const float*)d_in[9],  (const float*)d_in[15], (const float*)d_in[21] };
  const float* Bi[3]  = { (const float*)d_in[10], (const float*)d_in[16], (const float*)d_in[22] };
  const float* Gm[3]  = { (const float*)d_in[23], (const float*)d_in[25], (const float*)d_in[27] };
  const float* Bt_[3] = { (const float*)d_in[24], (const float*)d_in[26], (const float*)d_in[28] };
  const float* gateW  = (const float*)d_in[29];
  const float* gateB  = (const float*)d_in[30];
  const float* fcW    = (const float*)d_in[31];
  const float* fcB    = (const float*)d_in[32];

  const int Nn = in_sizes[4];            // 50000
  const int E  = in_sizes[1];            // 800000
  const int ET = E + Nn;
  const int Kdim[3]   = { 128, 128, 32 };
  const int Cout[3]   = { 128, 32, 32 };
  const int concat[3] = { 1, 0, 0 };

  // workspace layout
  char* ws = (char*)d_ws;
  size_t off = 0;
  auto take = [&](size_t bytes) -> void* {
    void* p = ws + off;
    off += (bytes + 255) & ~(size_t)255;
    return p;
  };
  unsigned short* xb  = (unsigned short*)take((size_t)Nn * HC * 2);
  unsigned short* wt  = (unsigned short*)take((size_t)128 * 128 * 2);
  float* h    = (float*)take((size_t)Nn * HC * 4);
  float* accb = (float*)take((size_t)Nn * HC * 4);
  float* xn   = (float*)take((size_t)Nn * HC * 4);
  float* asrc = (float*)take((size_t)Nn * HEADS * 4);
  float* adst = (float*)take((size_t)Nn * HEADS * 4);
  float* exb  = (float*)take((size_t)ET * HEADS * 4);   // alpha then ex
  unsigned* mx = (unsigned*)take((size_t)Nn * HEADS * 4);
  float* den  = (float*)take((size_t)Nn * HEADS * 4);
  float* sum16 = (float*)take(16 * 4);
  float* veff  = (float*)take(64 * 4);
  float* aloop = (float*)take(4 * 4);
  float* bnsum = (float*)take(128 * 4);
  float* bnsq  = (float*)take(128 * 4);
  float* bnsc  = (float*)take(128 * 4);
  float* bnsh  = (float*)take(128 * 4);
  float* gate  = (float*)take((size_t)Nn * 4);
  float* wexp  = (float*)take((size_t)Nn * 4);
  unsigned* gmax = (unsigned*)take(NG * 4);
  float* gden  = (float*)take(NG * 4);
  float* pooled = (float*)take(NG * HID * 4);
  float* y = h;                          // alias: h is dead once aggregation done

  auto blk = [](int n, int b) { return (n + b - 1) / b; };

  // edge_attr column mean (for self-loop fill)
  fill_f32<<<1, 16, 0, stream>>>(sum16, 0.f, 16);
  ea_sum_kernel<<<512, 256, 0, stream>>>(eattr, sum16, E);

  // initial input -> bf16
  cvt_bf16_kernel<<<blk(Nn * HC, 256), 256, 0, stream>>>(x_in, xb, Nn * HC);

  for (int l = 0; l < 3; ++l) {
    const int K = Kdim[l], C = Cout[l];

    layer_prep_kernel<<<1, 64, 0, stream>>>(sum16, Wev[l], Ae[l], veff, aloop, 1.0f / (float)E);
    transpose_w_kernel<<<blk(128 * K, 256), 256, 0, stream>>>(W[l], wt, K);

    // h = x @ W  (WMMA bf16 -> f32)
    {
      int waves = blk(Nn, 16);
      int threads = waves * 32;
      gemm_wmma_kernel<<<blk(threads, 256), 256, 0, stream>>>(xb, wt, h, Nn, K);
    }

    attn_node_kernel<<<blk(Nn * HEADS, 256), 256, 0, stream>>>(h, As[l], Ad[l], asrc, adst, Nn);

    fill_u32<<<blk(Nn * HEADS, 256), 256, 0, stream>>>(mx, ENC_NEG_INF, Nn * HEADS);
    fill_f32<<<blk(Nn * HEADS, 256), 256, 0, stream>>>(den, 0.f, Nn * HEADS);
    fill_f32<<<blk(Nn * HC, 256), 256, 0, stream>>>(accb, 0.f, Nn * HC);

    edge_alpha_kernel<<<blk(ET, 256), 256, 0, stream>>>(src, dst, eattr, veff, aloop,
                                                        asrc, adst, exb, mx, E, Nn);
    edge_exp_kernel<<<blk(ET, 256), 256, 0, stream>>>(dst, exb, mx, den, E, Nn);
    edge_aggr_kernel<<<blk(ET, 8), 256, 0, stream>>>(src, dst, exb, den, h, accb, E, Nn);

    fill_f32<<<1, 128, 0, stream>>>(bnsum, 0.f, 128);
    fill_f32<<<1, 128, 0, stream>>>(bnsq, 0.f, 128);
    post_y_stats_kernel<<<1024, 128, 0, stream>>>(accb, Bi[l], y, bnsum, bnsq, Nn, concat[l]);
    bn_finalize_kernel<<<1, 128, 0, stream>>>(bnsum, bnsq, Gm[l], Bt_[l], bnsc, bnsh, Nn, C);
    bn_elu_cvt_kernel<<<blk(Nn * C, 256), 256, 0, stream>>>(y, bnsc, bnsh, xn, xb, Nn, C);
  }

  // attentional pooling over graphs + final FC
  fill_u32<<<1, NG, 0, stream>>>(gmax, ENC_NEG_INF, NG);
  fill_f32<<<1, NG, 0, stream>>>(gden, 0.f, NG);
  fill_f32<<<blk(NG * HID, 256), 256, 0, stream>>>(pooled, 0.f, NG * HID);

  gate_max_kernel<<<blk(Nn, 256), 256, 0, stream>>>(xn, gateW, gateB, batch, gate, gmax, Nn);
  gate_exp_kernel<<<blk(Nn, 256), 256, 0, stream>>>(gate, batch, gmax, wexp, gden, Nn);
  pool_kernel<<<blk(Nn * HID, 256), 256, 0, stream>>>(xn, wexp, gden, batch, pooled, Nn);
  fc_kernel<<<1, 128, 0, stream>>>(pooled, fcW, fcB, (float*)d_out);
}
